// learn_SLIC_calc_v2_48095043780760
// MI455X (gfx1250) — matrix operand order repaired and checked by reading, compiled, not run on hardware
//
#include <hip/hip_runtime.h>
#include <math.h>

typedef __attribute__((ext_vector_type(16))) _Float16 v16h;
typedef __attribute__((ext_vector_type(8)))  float    v8f;

#define KNB 6

#define WMMA16(A, B, C) \
  __builtin_amdgcn_wmma_f32_16x16x32_f16(false, (A), false, (B), (short)0, (C), false, false)

// ---------------------------------------------------------------------------
// Prep: fold eval-mode BatchNorm into layer-1 weights/biases.
// ---------------------------------------------------------------------------
__global__ void slic_prep(
    const float* __restrict__ Wf1, const float* __restrict__ bf1,
    const float* __restrict__ gf,  const float* __restrict__ btf,
    const float* __restrict__ mf,  const float* __restrict__ vf,
    const float* __restrict__ Wx1, const float* __restrict__ bx1,
    const float* __restrict__ gx,  const float* __restrict__ btx,
    const float* __restrict__ mx,  const float* __restrict__ vx,
    const float* __restrict__ Wm1, const float* __restrict__ bm1,
    const float* __restrict__ gm,  const float* __restrict__ btm,
    const float* __restrict__ mm,  const float* __restrict__ vm,
    float* __restrict__ W1f,  float* __restrict__ b1f,
    float* __restrict__ Wx1f, float* __restrict__ bx1f,
    float* __restrict__ Wm1f, float* __restrict__ bm1f)
{
  const int t = threadIdx.x;
  for (int i = t; i < 32 * 64; i += blockDim.x) {
    int o = i >> 6;
    float sf = gf[o] * rsqrtf(vf[o] + 1e-5f);
    float sm = gm[o] * rsqrtf(vm[o] + 1e-5f);
    W1f[i]  = Wf1[i] * sf;
    Wm1f[i] = Wm1[i] * sm;
  }
  for (int o = t; o < 32; o += blockDim.x) {
    float sf = gf[o] * rsqrtf(vf[o] + 1e-5f);
    float sm = gm[o] * rsqrtf(vm[o] + 1e-5f);
    b1f[o]  = (bf1[o] - mf[o]) * sf + btf[o];
    bm1f[o] = (bm1[o] - mm[o]) * sm + btm[o];
  }
  for (int i = t; i < 16 * 3; i += blockDim.x) {
    int o = i / 3;
    float sx = gx[o] * rsqrtf(vx[o] + 1e-5f);
    Wx1f[i] = Wx1[i] * sx;
  }
  for (int o = t; o < 16; o += blockDim.x) {
    float sx = gx[o] * rsqrtf(vx[o] + 1e-5f);
    bx1f[o] = (bx1[o] - mx[o]) * sx + btx[o];
  }
}

// Load a 16-half A-fragment (32 bytes) from LDS (16B aligned, 48B lane stride).
__device__ __forceinline__ v16h load_frag(const unsigned short* p) {
  union { uint4 u4[2]; v16h v; } t;
  const uint4* q = (const uint4*)p;
  t.u4[0] = q[0];
  t.u4[1] = q[1];
  return t.v;
}

// ---------------------------------------------------------------------------
// Main fused kernel (wave32): each wave owns 16 points (lanes l and l+16 share
// point r = l&15). Matmuls computed transposed: D(M=out-ch, N=point-row) =
// A(weights) x B(data^T). Weight A-fragments are pre-swizzled into LDS once
// per block (shared by all 4 waves) to keep VGPR count low -> more waves in
// flight for the latency-bound gather/atomic phases.
// ---------------------------------------------------------------------------
__launch_bounds__(128, 4)
__global__ void slic_main(
    const float* __restrict__ sp_fea, const float* __restrict__ sp_xyz,
    const float* __restrict__ o_p_fea, const float* __restrict__ p_xyz,
    const int*   __restrict__ c2p_idx,
    const float* __restrict__ W1f,  const float* __restrict__ b1f,
    const float* __restrict__ Wf2,  const float* __restrict__ bf2,
    const float* __restrict__ Wx1f, const float* __restrict__ bx1f,
    const float* __restrict__ Wx2,  const float* __restrict__ bx2,
    const float* __restrict__ Wm1f, const float* __restrict__ bm1f,
    const float* __restrict__ Wm2,  const float* __restrict__ bm2,
    float* __restrict__ out_fea, float* __restrict__ out_xyz,
    float* __restrict__ sp_sum, int N)
{
  // frag f, lane l: 16 f16 elements at stride 48B (bank spread, 16B aligned)
  __shared__ __align__(16) unsigned short s_frag[6][32][24];
  __shared__ float s_b1f[32], s_bf2[16], s_bx1[16], s_bx2[16], s_wx1[48];
  __shared__ float s_w[4][16][KNB];
  __shared__ int   s_id[4][16][KNB];

  const int tid  = threadIdx.x;
  const int lane = tid & 31;
  const int wv   = tid >> 5;
  const int r    = lane & 15;     // point slot within wave
  const int hi   = lane >> 4;     // half-wave id
  const int p0   = (blockIdx.x * 4 + wv) * 16;
  const int row  = min(p0 + r, N - 1);

  // ---- block-cooperative staging of k-loop weight fragments into LDS ----
  // fragments: f=0..3 -> Af1[h=f>>1][q=f&1]; f=4 -> Af2; f=5 -> Ax2 (zero-pad)
  for (int i = tid; i < 6 * 32 * 16; i += 128) {
    int f = i >> 9;
    int l = (i >> 4) & 31;
    int e = i & 15;
    int rr = l & 15, hh = l >> 4;
    int K = hh * 16 + e;   // A-layout K seq (contiguous)
    float val;
    if (f < 4) {
      int h = f >> 1, q = f & 1;
      val = W1f[(h * 16 + rr) * 64 + q * 32 + K];
    } else if (f == 4) {
      val = Wf2[rr * 32 + K];
    } else {
      val = (K < 16) ? Wx2[rr * 16 + K] : 0.0f;
    }
    union { _Float16 h16; unsigned short u; } cv;
    cv.h16 = (_Float16)val;
    s_frag[f][l][e] = cv.u;
  }
  for (int i = tid; i < 32; i += 128) s_b1f[i] = b1f[i];
  for (int i = tid; i < 48; i += 128) s_wx1[i] = Wx1f[i];
  if (tid < 16) {
    s_bf2[tid] = bf2[tid];
    s_bx1[tid] = bx1f[tid];
    s_bx2[tid] = bx2[tid];
  }
  __syncthreads();

  // ---- per-point data: lane holds 16 contiguous channels per K-chunk ----
  float opf[2][16];
#pragma unroll
  for (int q = 0; q < 2; ++q) {
    const float4* p4 = (const float4*)(o_p_fea + (size_t)row * 64 + q * 32 + hi * 16);
#pragma unroll
    for (int t = 0; t < 4; ++t) {
      float4 v = p4[t];
      opf[q][4 * t + 0] = v.x; opf[q][4 * t + 1] = v.y;
      opf[q][4 * t + 2] = v.z; opf[q][4 * t + 3] = v.w;
    }
  }
  const float px0 = p_xyz[row * 3 + 0];
  const float px1 = p_xyz[row * 3 + 1];
  const float px2 = p_xyz[row * 3 + 2];
  int idx6[KNB];
#pragma unroll
  for (int k = 0; k < KNB; ++k) idx6[k] = c2p_idx[row * KNB + k];

  const v8f vzero = {};

  // ---- per-point MLP -> normalized p_fea tile (M=channel, N=point) ----
  // (mask-MLP weights used exactly once -> transient direct global loads)
  float pf[8];
  {
    v8f accM0 = vzero, accM1 = vzero;
#pragma unroll
    for (int q = 0; q < 2; ++q) {
      v16h Am0, Am1v;
#pragma unroll
      for (int e = 0; e < 16; ++e) {
        int K = q * 32 + hi * 16 + e;
        Am0[e]  = (_Float16)Wm1f[r * 64 + K];
        Am1v[e] = (_Float16)Wm1f[(16 + r) * 64 + K];
      }
      v16h B;
#pragma unroll
      for (int e = 0; e < 16; ++e) B[e] = (_Float16)opf[q][e];
      accM0 = WMMA16(Am0, B, accM0);
      accM1 = WMMA16(Am1v, B, accM1);
    }
    float h0[8], h1[8];
#pragma unroll
    for (int j = 0; j < 8; ++j) {
      h0[j] = fmaxf(accM0[j] + bm1f[j + 8 * hi], 0.0f);
      h1[j] = fmaxf(accM1[j] + bm1f[16 + j + 8 * hi], 0.0f);
    }
    v16h B2;  // layer-2 B fragment via half-wave swap
#pragma unroll
    for (int j = 0; j < 8; ++j) {
      float s0 = __shfl_xor(h0[j], 16);
      float s1 = __shfl_xor(h1[j], 16);
      B2[j]     = (_Float16)(hi ? s1 : h0[j]);
      B2[8 + j] = (_Float16)(hi ? h1[j] : s0);
    }
    v16h Am2;
#pragma unroll
    for (int e = 0; e < 16; ++e) {
      int K = hi * 16 + e;
      Am2[e] = (_Float16)Wm2[r * 32 + K];
    }
    v8f accP = WMMA16(Am2, B2, vzero);
    float ss = 0.0f;
#pragma unroll
    for (int j = 0; j < 8; ++j) { pf[j] = accP[j] + bm2[j + 8 * hi]; ss += pf[j] * pf[j]; }
    ss += __shfl_xor(ss, 16);
    float sc = 1.0f / fmaxf(sqrtf(ss), 1e-12f);
#pragma unroll
    for (int j = 0; j < 8; ++j) pf[j] *= sc;
  }

  // ---- six neighbor tiles ----
  float bw[KNB];
  for (int k = 0; k < KNB; ++k) {
    const int id = idx6[k];
    // feature branch: c2p_fea = sp_fea[id] - o_p_fea[row]
    v8f a0 = vzero, a1 = vzero;
#pragma unroll
    for (int q = 0; q < 2; ++q) {
      const float4* sp4 = (const float4*)(sp_fea + (size_t)id * 64 + q * 32 + hi * 16);
      float f[16];
#pragma unroll
      for (int t = 0; t < 4; ++t) {
        float4 v = sp4[t];
        f[4 * t + 0] = v.x; f[4 * t + 1] = v.y;
        f[4 * t + 2] = v.z; f[4 * t + 3] = v.w;
      }
      v16h B;
#pragma unroll
      for (int e = 0; e < 16; ++e) B[e] = (_Float16)(f[e] - opf[q][e]);
      a0 = WMMA16(load_frag(&s_frag[q][lane][0]),     B, a0);
      a1 = WMMA16(load_frag(&s_frag[2 + q][lane][0]), B, a1);
    }
    float g0[8], g1[8];
#pragma unroll
    for (int j = 0; j < 8; ++j) {
      g0[j] = fmaxf(a0[j] + s_b1f[j + 8 * hi], 0.0f);
      g1[j] = fmaxf(a1[j] + s_b1f[16 + j + 8 * hi], 0.0f);
    }
    v16h B2;
#pragma unroll
    for (int j = 0; j < 8; ++j) {
      float s0 = __shfl_xor(g0[j], 16);
      float s1 = __shfl_xor(g1[j], 16);
      B2[j]     = (_Float16)(hi ? s1 : g0[j]);
      B2[8 + j] = (_Float16)(hi ? g1[j] : s0);
    }
    v8f wfacc = WMMA16(load_frag(&s_frag[4][lane][0]), B2, vzero);
    float dotf;
    {
      float wf[8], ss = 0.0f;
#pragma unroll
      for (int j = 0; j < 8; ++j) { wf[j] = wfacc[j] + s_bf2[j + 8 * hi]; ss += wf[j] * wf[j]; }
      ss += __shfl_xor(ss, 16);
      float sc = 1.0f / fmaxf(sqrtf(ss), 1e-12f);
      float d = 0.0f;
#pragma unroll
      for (int j = 0; j < 8; ++j) d += (wf[j] * sc) * pf[j];
      d += __shfl_xor(d, 16);
      dotf = d;
    }
    // xyz branch: layer-1 (K=3) on VALU directly in C/D layout, layer-2 WMMA
    float dx = sp_xyz[id * 3 + 0] - px0;
    float dy = sp_xyz[id * 3 + 1] - px1;
    float dz = sp_xyz[id * 3 + 2] - px2;
    float hx[8];
#pragma unroll
    for (int j = 0; j < 8; ++j) {
      int ch = j + 8 * hi;
      hx[j] = fmaxf(s_wx1[ch * 3 + 0] * dx + s_wx1[ch * 3 + 1] * dy +
                    s_wx1[ch * 3 + 2] * dz + s_bx1[ch], 0.0f);
    }
    v16h Bx;
#pragma unroll
    for (int j = 0; j < 8; ++j) {
      float sx = __shfl_xor(hx[j], 16);
      Bx[j]     = hi ? (_Float16)0.0f : (_Float16)hx[j];
      Bx[8 + j] = hi ? (_Float16)0.0f : (_Float16)sx;
    }
    v8f wxacc = WMMA16(load_frag(&s_frag[5][lane][0]), Bx, vzero);
    float dotx;
    {
      float wx[8], ss = 0.0f;
#pragma unroll
      for (int j = 0; j < 8; ++j) { wx[j] = wxacc[j] + s_bx2[j + 8 * hi]; ss += wx[j] * wx[j]; }
      ss += __shfl_xor(ss, 16);
      float sc = 1.0f / fmaxf(sqrtf(ss), 1e-12f);
      float d = 0.0f;
#pragma unroll
      for (int j = 0; j < 8; ++j) d += (wx[j] * sc) * pf[j];
      d += __shfl_xor(d, 16);
      dotx = d;
    }
    bw[k] = dotf * dotx;
  }

  // ---- softmax over the 6 candidates (lane-local) ----
  float mxv = bw[0];
#pragma unroll
  for (int k = 1; k < KNB; ++k) mxv = fmaxf(mxv, bw[k]);
  float sum = 0.0f;
#pragma unroll
  for (int k = 0; k < KNB; ++k) { bw[k] = __expf(bw[k] - mxv); sum += bw[k]; }
  float inv = 1.0f / sum;

  if (hi == 0) {
#pragma unroll
    for (int k = 0; k < KNB; ++k) {
      s_w[wv][r][k]  = bw[k] * inv;
      s_id[wv][r][k] = idx6[k];
    }
  }
  __syncthreads();

  // ---- scatter (segment sums) ----
  for (int k = 0; k < KNB; ++k) {
    // 16 points x 64 channels = 1024 adds, 32 lanes x 32 iters
    for (int i = 0; i < 32; ++i) {
      int flat = i * 32 + lane;
      int p = flat >> 6, c = flat & 63;
      if (p0 + p < N) {
        float w  = s_w[wv][p][k];
        int   id = s_id[wv][p][k];
        atomicAdd(&out_fea[(size_t)id * 64 + c], w * o_p_fea[(size_t)(p0 + p) * 64 + c]);
      }
    }
    // 16 points x (1 sum + 3 xyz) = 64 adds
    for (int i = 0; i < 2; ++i) {
      int flat = i * 32 + lane;
      int p = flat >> 2, v = flat & 3;
      if (p0 + p < N) {
        float w  = s_w[wv][p][k];
        int   id = s_id[wv][p][k];
        if (v == 0) atomicAdd(&sp_sum[id], w);
        else atomicAdd(&out_xyz[(size_t)id * 3 + (v - 1)],
                       w * p_xyz[(size_t)(p0 + p) * 3 + (v - 1)]);
      }
    }
  }
}

// ---------------------------------------------------------------------------
// Finalize: divide accumulated sums by (sp_sum + 1e-8), in place on d_out.
// ---------------------------------------------------------------------------
__global__ void slic_finalize(float* __restrict__ out, const float* __restrict__ sp_sum, int M)
{
  int i = blockIdx.x * blockDim.x + threadIdx.x;
  int nf = M * 64, total = M * 67;
  if (i >= total) return;
  int m = (i < nf) ? (i >> 6) : ((i - nf) / 3);
  out[i] = out[i] / (sp_sum[m] + 1e-8f);
}

extern "C" void kernel_launch(void* const* d_in, const int* in_sizes, int n_in,
                              void* d_out, int out_size, void* d_ws, size_t ws_size,
                              hipStream_t stream) {
  (void)n_in; (void)ws_size;
  const float* sp_fea  = (const float*)d_in[0];
  const float* sp_xyz  = (const float*)d_in[1];
  const float* o_p_fea = (const float*)d_in[2];
  const float* p_xyz   = (const float*)d_in[3];
  const int*   c2p_abs = (const int*)d_in[4];   // gather indices (== scatter indices here)
  const float* Wf1 = (const float*)d_in[9];
  const float* bf1 = (const float*)d_in[10];
  const float* gf  = (const float*)d_in[11];
  const float* btf = (const float*)d_in[12];
  const float* mf  = (const float*)d_in[13];
  const float* vf  = (const float*)d_in[14];
  const float* Wf2 = (const float*)d_in[15];
  const float* bf2 = (const float*)d_in[16];
  const float* Wx1 = (const float*)d_in[17];
  const float* bx1 = (const float*)d_in[18];
  const float* gx  = (const float*)d_in[19];
  const float* btx = (const float*)d_in[20];
  const float* mx  = (const float*)d_in[21];
  const float* vx  = (const float*)d_in[22];
  const float* Wx2 = (const float*)d_in[23];
  const float* bx2 = (const float*)d_in[24];
  const float* Wm1 = (const float*)d_in[25];
  const float* bm1 = (const float*)d_in[26];
  const float* gm  = (const float*)d_in[27];
  const float* btm = (const float*)d_in[28];
  const float* mm  = (const float*)d_in[29];
  const float* vm  = (const float*)d_in[30];
  const float* Wm2 = (const float*)d_in[31];
  const float* bm2 = (const float*)d_in[32];

  const int M = in_sizes[0] / 64;   // 5000
  const int N = in_sizes[2] / 64;   // 300000

  float* out     = (float*)d_out;
  float* out_fea = out;                      // [M,64]
  float* out_xyz = out + (size_t)M * 64;     // [M,3]

  float* ws     = (float*)d_ws;
  float* sp_sum = ws;                 // M
  float* W1f    = ws + M;             // 2048
  float* b1f    = W1f + 2048;         // 32
  float* Wx1f   = b1f + 32;           // 48
  float* bx1f   = Wx1f + 48;          // 16
  float* Wm1f   = bx1f + 16;          // 2048
  float* bm1f   = Wm1f + 2048;        // 32

  hipMemsetAsync(d_out, 0, (size_t)out_size * sizeof(float), stream);
  hipMemsetAsync(sp_sum, 0, (size_t)M * sizeof(float), stream);

  slic_prep<<<1, 256, 0, stream>>>(Wf1, bf1, gf, btf, mf, vf,
                                   Wx1, bx1, gx, btx, mx, vx,
                                   Wm1, bm1, gm, btm, mm, vm,
                                   W1f, b1f, Wx1f, bx1f, Wm1f, bm1f);

  int blocks = (N + 63) / 64;   // 4 waves/block, 16 points/wave
  slic_main<<<blocks, 128, 0, stream>>>(sp_fea, sp_xyz, o_p_fea, p_xyz, c2p_abs,
                                        W1f, b1f, Wf2, bf2, Wx1f, bx1f, Wx2, bx2,
                                        Wm1f, bm1f, Wm2, bm2,
                                        out_fea, out_xyz, sp_sum, N);

  slic_finalize<<<(M * 67 + 255) / 256, 256, 0, stream>>>(out, sp_sum, M);
}
